// MultiHeadAttention_48825188221163
// MI455X (gfx1250) — matrix operand order repaired and checked by reading, not compiled, not run on hardware
//
#include <hip/hip_runtime.h>

#define D_MODEL 768
#define NUM_HEADS 12
#define D_K 64
#define SEQ 2048
#define BATCH 2
#define NROWS (BATCH * SEQ)   // 4096

typedef __attribute__((ext_vector_type(16))) _Float16 v16h;
typedef __attribute__((ext_vector_type(8)))  _Float16 v8h;
typedef __attribute__((ext_vector_type(4)))  _Float16 v4h;
typedef __attribute__((ext_vector_type(8)))  float    v8f;
typedef __attribute__((ext_vector_type(4)))  float    v4f;

union Frag16 {
    v16h v;
    v8h  h[2];
};

__device__ __forceinline__ v8f wmma_f16(v16h a, v16h b, v8f c) {
    // D = A(16x32 f16) * B(32x16 f16) + C(16x16 f32)
    return __builtin_amdgcn_wmma_f32_16x16x32_f16(false, a, false, b, (short)0, c, false, false);
}

// ---------------- prep kernels ----------------

// 4-wide f32 -> f16 convert (n must be a multiple of 4; it is: 3145728)
__global__ void cvt_f32_to_f16_v4(const float* __restrict__ in, _Float16* __restrict__ out, int n4) {
    int i = blockIdx.x * blockDim.x + threadIdx.x;
    if (i < n4) {
        v4f x = ((const v4f*)in)[i];
        v4h h;
        h.x = (_Float16)x.x; h.y = (_Float16)x.y;
        h.z = (_Float16)x.z; h.w = (_Float16)x.w;
        ((v4h*)out)[i] = h;
    }
}

// W is [K=768][N=768] row-major; produce Wt[n][k] f16 so B-fragments are contiguous in K.
__global__ void wtrans(const float* __restrict__ W, _Float16* __restrict__ Wt) {
    int i = blockIdx.x * blockDim.x + threadIdx.x;
    if (i < D_MODEL * D_MODEL) {
        int k = i / D_MODEL, n = i - k * D_MODEL;
        Wt[n * D_MODEL + k] = (_Float16)W[i];
    }
}

// ---------------- WMMA GEMM: Y = X @ W + bias ----------------
// One wave per block computes a 16x64 tile. X: [NROWS][768] f16 row-major,
// Wt: [768(N)][768(K)] f16. mode 0: f16 out [B,H,S,dk]; mode 1: f16 out
// [B,H,dk,S] (transposed V); mode 2: f32 out row-major [NROWS][768] (NT stores).
__global__ __launch_bounds__(32)
void gemm16x64(const _Float16* __restrict__ X,
               const _Float16* __restrict__ Wt,
               const float* __restrict__ bias,
               _Float16* __restrict__ out_h,
               float* __restrict__ out_f,
               int mode)
{
    const int lane = threadIdx.x;
    const int hl = lane >> 4, l15 = lane & 15;
    const int m0 = blockIdx.y << 4;
    const int n0 = blockIdx.x << 6;

    v8f acc[4] = {};
    const _Float16* xrow = X + (size_t)(m0 + l15) * D_MODEL;

    for (int k0 = 0; k0 < D_MODEL; k0 += 32) {
        Frag16 a;
        a.h[0] = *(const v8h*)(xrow + k0 + hl * 16);       // K = hl*16 + j
        a.h[1] = *(const v8h*)(xrow + k0 + hl * 16 + 8);   // K = hl*16 + 8 + j
#pragma unroll
        for (int t = 0; t < 4; ++t) {
            const _Float16* wr = Wt + (size_t)(n0 + t * 16 + l15) * D_MODEL + k0 + hl * 16;
            Frag16 b;
            b.h[0] = *(const v8h*)(wr);      // K = hl*16 + j, N = l15
            b.h[1] = *(const v8h*)(wr + 8);
            acc[t] = wmma_f16(a.v, b.v, acc[t]);
        }
    }

#pragma unroll
    for (int t = 0; t < 4; ++t) {
        const int col = n0 + t * 16 + l15;
        const float bv = bias[col];
#pragma unroll
        for (int g = 0; g < 8; ++g) {
            const int row = m0 + g + hl * 8;   // C layout: M = g + 8*halfwave
            const float val = acc[t][g] + bv;
            if (mode == 2) {
                __builtin_nontemporal_store(val, out_f + (size_t)row * D_MODEL + col);
            } else {
                const int bb = row >> 11;       // / SEQ
                const int s  = row & (SEQ - 1);
                const int h  = col >> 6;        // / D_K
                const int d  = col & (D_K - 1);
                const size_t bh = (size_t)(bb * NUM_HEADS + h);
                if (mode == 0)
                    out_h[(bh * SEQ + s) * D_K + d] = (_Float16)val;   // [B,H,S,dk]
                else
                    out_h[(bh * D_K + d) * SEQ + s] = (_Float16)val;   // [B,H,dk,S]
            }
        }
    }
}

// ---------------- fused attention ----------------
// grid: (SEQ/16, B*H), block 256 (8 waves). Dynamic LDS:
//   sc  : 16 x 2048 f32 scores (128 KB)   (reused for split-K partials)
//   pm  : 16 x 2048 f16 normalized probs (64 KB)
__global__ __launch_bounds__(256)
void attn_fused(const _Float16* __restrict__ Qh,   // [B*H][S][64]
                const _Float16* __restrict__ Kh,   // [B*H][S][64]
                const _Float16* __restrict__ Vt,   // [B*H][64][S]
                float* __restrict__ wout,          // [B*H][S][S]
                _Float16* __restrict__ ctx)        // [B][S][768] f16
{
    const int bh  = blockIdx.y;
    const int m0  = blockIdx.x << 4;
    const int tid = threadIdx.x;
    const int w = tid >> 5;
    const int lane = tid & 31;
    const int hl = lane >> 4, l15 = lane & 15;

    extern __shared__ char smem[];
    float*    sc = (float*)smem;                          // [16][SEQ]
    _Float16* pm = (_Float16*)(smem + 16 * SEQ * 4);      // [16][SEQ]

    const _Float16* Q = Qh + (size_t)bh * SEQ * D_K;
    const _Float16* K = Kh + (size_t)bh * SEQ * D_K;
    const _Float16* V = Vt + (size_t)bh * D_K * SEQ;

    // A-panel (Q tile 16x64) as two K=32 fragments, kept in registers.
    Frag16 aq[2];
    {
        const _Float16* qr = Q + (size_t)(m0 + l15) * D_K;
#pragma unroll
        for (int kk = 0; kk < 2; ++kk) {
            aq[kk].h[0] = *(const v8h*)(qr + kk * 32 + hl * 16);
            aq[kk].h[1] = *(const v8h*)(qr + kk * 32 + hl * 16 + 8);
        }
    }

    // Phase 1: scores = Q K^T * 1/8 ; wave w owns 256 columns.
#pragma unroll 1
    for (int t = 0; t < 16; ++t) {
        const int n0 = (w << 8) + (t << 4);
        v8f acc = {};
#pragma unroll
        for (int kk = 0; kk < 2; ++kk) {
            // B[k][n] = K[n0+n][k]  -> contiguous in K along a Kh row
            const _Float16* kr = K + (size_t)(n0 + l15) * D_K + kk * 32 + hl * 16;
            Frag16 b;
            b.h[0] = *(const v8h*)(kr);
            b.h[1] = *(const v8h*)(kr + 8);
            acc = wmma_f16(aq[kk].v, b.v, acc);
        }
#pragma unroll
        for (int g = 0; g < 8; ++g)
            sc[(g + hl * 8) * SEQ + n0 + l15] = acc[g] * 0.125f;
    }
    __syncthreads();

    // Phase 2: softmax; wave w owns rows 2w, 2w+1. 4-wide passes.
#pragma unroll 1
    for (int rr = 0; rr < 2; ++rr) {
        const int row = (w << 1) + rr;
        v4f* srow4 = (v4f*)(sc + row * SEQ);
        float mx = -3.0e38f;
        for (int i = lane; i < SEQ / 4; i += 32) {
            v4f x = srow4[i];
            mx = fmaxf(mx, fmaxf(fmaxf(x.x, x.y), fmaxf(x.z, x.w)));
        }
#pragma unroll
        for (int off = 16; off > 0; off >>= 1) mx = fmaxf(mx, __shfl_xor(mx, off, 32));
        float sum = 0.f;
        for (int i = lane; i < SEQ / 4; i += 32) {
            v4f x = srow4[i];
            v4f e;
            e.x = __expf(x.x - mx); e.y = __expf(x.y - mx);
            e.z = __expf(x.z - mx); e.w = __expf(x.w - mx);
            srow4[i] = e;
            sum += (e.x + e.y) + (e.z + e.w);
        }
#pragma unroll
        for (int off = 16; off > 0; off >>= 1) sum += __shfl_xor(sum, off, 32);
        const float inv = 1.0f / sum;
        v4f* wrow4 = (v4f*)(wout + ((size_t)bh * SEQ + (size_t)(m0 + row)) * SEQ);
        v4h* prow4 = (v4h*)(pm + row * SEQ);
        for (int i = lane; i < SEQ / 4; i += 32) {
            v4f e = srow4[i];
            v4f wv;
            wv.x = e.x * inv; wv.y = e.y * inv;
            wv.z = e.z * inv; wv.w = e.w * inv;
            __builtin_nontemporal_store(wv, wrow4 + i);   // f32 weights, streamed (NT)
            v4h ph;
            ph.x = (_Float16)wv.x; ph.y = (_Float16)wv.y;
            ph.z = (_Float16)wv.z; ph.w = (_Float16)wv.w;
            prow4[i] = ph;                                // normalized probs for P@V
        }
    }
    __syncthreads();

    // Phase 3: ctx partial; wave w covers K-range [w*256, w*256+256).
    v8f acc[4] = {};
    const _Float16* prow = pm + (size_t)l15 * SEQ;   // A row M = l15
#pragma unroll 1
    for (int ks = 0; ks < 8; ++ks) {
        const int k0 = (w << 8) + (ks << 5);
        Frag16 a;
        a.h[0] = *(const v8h*)(prow + k0 + hl * 16);
        a.h[1] = *(const v8h*)(prow + k0 + hl * 16 + 8);
#pragma unroll
        for (int t = 0; t < 4; ++t) {
            // B[k][n] = V[k][n] = Vt[n][k] -> contiguous in K along a Vt row
            const _Float16* vr = V + (size_t)(t * 16 + l15) * SEQ + k0 + hl * 16;
            Frag16 b;
            b.h[0] = *(const v8h*)(vr);
            b.h[1] = *(const v8h*)(vr + 8);
            acc[t] = wmma_f16(a.v, b.v, acc[t]);
        }
    }
    // split-K partials -> sc (free after softmax barrier)
    float* part = sc + (w << 10);
#pragma unroll
    for (int t = 0; t < 4; ++t)
#pragma unroll
        for (int g = 0; g < 8; ++g)
            part[(g + hl * 8) * D_K + t * 16 + l15] = acc[t][g];
    __syncthreads();

    const int bb = bh / NUM_HEADS;
    const int h  = bh % NUM_HEADS;
    for (int o = tid; o < 16 * D_K; o += 256) {
        float s = 0.f;
#pragma unroll
        for (int ww = 0; ww < 8; ++ww) s += sc[(ww << 10) + o];
        const int row = o >> 6, col = o & (D_K - 1);
        ctx[(size_t)(bb * SEQ + m0 + row) * D_MODEL + h * D_K + col] = (_Float16)s;
    }
}

// ---------------- launch ----------------

extern "C" void kernel_launch(void* const* d_in, const int* in_sizes, int n_in,
                              void* d_out, int out_size, void* d_ws, size_t ws_size,
                              hipStream_t stream) {
    (void)in_sizes; (void)n_in; (void)out_size; (void)ws_size;

    const float* q   = (const float*)d_in[0];
    const float* k   = (const float*)d_in[1];
    const float* v   = (const float*)d_in[2];
    const float* w_q = (const float*)d_in[3];
    const float* b_q = (const float*)d_in[4];
    const float* w_k = (const float*)d_in[5];
    const float* b_k = (const float*)d_in[6];
    const float* w_v = (const float*)d_in[7];
    const float* b_v = (const float*)d_in[8];
    const float* w_o = (const float*)d_in[9];
    const float* b_o = (const float*)d_in[10];

    float* out_f   = (float*)d_out;                          // [4096][768]
    float* weights = out_f + (size_t)NROWS * D_MODEL;        // [24][2048][2048]

    char* ws = (char*)d_ws;
    size_t off = 0;
    auto alloc = [&](size_t bytes) -> void* {
        void* p = ws + off;
        off += (bytes + 255) & ~(size_t)255;
        return p;
    };
    const size_t NX = (size_t)NROWS * D_MODEL;         // 3145728
    const size_t NW = (size_t)D_MODEL * D_MODEL;       // 589824
    _Float16* qx  = (_Float16*)alloc(NX * 2);
    _Float16* kx  = (_Float16*)alloc(NX * 2);
    _Float16* vx  = (_Float16*)alloc(NX * 2);
    _Float16* wqt = (_Float16*)alloc(NW * 2);
    _Float16* wkt = (_Float16*)alloc(NW * 2);
    _Float16* wvt = (_Float16*)alloc(NW * 2);
    _Float16* wot = (_Float16*)alloc(NW * 2);
    _Float16* Qh  = (_Float16*)alloc(NX * 2);
    _Float16* Kh  = (_Float16*)alloc(NX * 2);
    _Float16* Vt  = (_Float16*)alloc(NX * 2);
    _Float16* ctx = (_Float16*)alloc(NX * 2);

    {
        const int n4 = (int)(NX / 4);
        dim3 g((n4 + 255) / 256);
        cvt_f32_to_f16_v4<<<g, 256, 0, stream>>>(q, qx, n4);
        cvt_f32_to_f16_v4<<<g, 256, 0, stream>>>(k, kx, n4);
        cvt_f32_to_f16_v4<<<g, 256, 0, stream>>>(v, vx, n4);
    }
    {
        const int n = (int)NW;
        dim3 g((n + 255) / 256);
        wtrans<<<g, 256, 0, stream>>>(w_q, wqt);
        wtrans<<<g, 256, 0, stream>>>(w_k, wkt);
        wtrans<<<g, 256, 0, stream>>>(w_v, wvt);
        wtrans<<<g, 256, 0, stream>>>(w_o, wot);
    }

    dim3 gg(D_MODEL / 64, NROWS / 16);  // 12 x 256
    gemm16x64<<<gg, 32, 0, stream>>>(qx, wqt, b_q, Qh, nullptr, 0);
    gemm16x64<<<gg, 32, 0, stream>>>(kx, wkt, b_k, Kh, nullptr, 0);
    gemm16x64<<<gg, 32, 0, stream>>>(vx, wvt, b_v, Vt, nullptr, 1);

    dim3 ga(SEQ / 16, BATCH * NUM_HEADS); // 128 x 24
    const size_t smem = (size_t)16 * SEQ * 4 + (size_t)16 * SEQ * 2; // 192 KB
    attn_fused<<<ga, 256, smem, stream>>>(Qh, Kh, Vt, weights, ctx);

    gemm16x64<<<gg, 32, 0, stream>>>(ctx, wot, b_o, nullptr, out_f, 2);
}
